// TinyRNN_9371618639959
// MI455X (gfx1250) — matrix operand order repaired and checked by reading, not compiled, not run on hardware
//
#include <hip/hip_runtime.h>
#include <math.h>

typedef __bf16 bf16_t;
typedef __attribute__((ext_vector_type(16))) __bf16 v16bf;
typedef __attribute__((ext_vector_type(8)))  float  v8f;
typedef unsigned int u32;
typedef __attribute__((ext_vector_type(4))) u32 u32x4;
typedef __attribute__((ext_vector_type(8))) int i32x8;
typedef __attribute__((ext_vector_type(4))) int i32x4;

#define S_LEN 2048
#define NB    32
#define NH    256
#define HPITCH 264   // bf16/LDS row pitch: 528B stride -> 16 lanes hit all 64 banks

__device__ __forceinline__ v8f wmma_bf16(v16bf a, v16bf b, v8f c) {
  // D = A(16x32 bf16) * B(32x16 bf16) + C(16x16 f32)
  return __builtin_amdgcn_wmma_f32_16x16x32_bf16(false, a, false, b, (short)0, c,
                                                 false, false);
}

// ---- WMMA fragment loaders (lane layouts per CDNA5 ISA 7.12.2) -------------

__device__ __forceinline__ v16bf lds_afrag(const bf16_t* hbuf, int mbase, int kc, int lane) {
  const int row = mbase + (lane & 15);
  const int kb  = kc * 32 + ((lane & 16) ? 8 : 0);
  const bf16_t* p = hbuf + row * HPITCH + kb;
  v16bf a;
#pragma unroll
  for (int j = 0; j < 8; ++j) { a[j] = p[j]; a[j + 8] = p[j + 16]; }
  return a;
}

__device__ __forceinline__ v16bf gbl_afrag_f32(const float* rowptr, int kc, int lane) {
  const int kb = kc * 32 + ((lane & 16) ? 8 : 0);
  v16bf a;
#pragma unroll
  for (int j = 0; j < 8; ++j) {
    a[j]     = (bf16_t)rowptr[kb + j];
    a[j + 8] = (bf16_t)rowptr[kb + 16 + j];
  }
  return a;
}

__device__ __forceinline__ v16bf gbl_bfrag(const float* W, int col, int kc, int lane) {
  const int kb = kc * 32 + ((lane & 16) ? 8 : 0);
  v16bf b;
#pragma unroll
  for (int j = 0; j < 16; ++j) b[j] = (bf16_t)W[(size_t)(kb + (j & 7) + ((j >> 3) << 4)) * NH + col];
  return b;
}

// ---- TDM: DMA one 32KB xW step-tile (8192 f32, contiguous) into LDS --------
// D# per CDNA5 ISA ch.8: group0 = {flags/count, lds_addr, global_addr lo/hi+type},
// group1 = {mask/dsize, dims/tile dims/strides}; groups 2&3 zero (<=2D tensor).
__device__ __forceinline__ void tdm_load_tile(const float* gsrc, void* lds_dst) {
  const unsigned long long ga = (unsigned long long)(uintptr_t)gsrc;
  const u32 lds_off = (u32)(uintptr_t)lds_dst;      // LDS aperture: offset = addr[31:0]
  u32x4 g0;
  g0[0] = 1u;                                       // count=1 (valid), user mode
  g0[1] = lds_off;                                  // lds_addr (bytes)
  g0[2] = (u32)(ga & 0xffffffffu);                  // global_addr[31:0]
  g0[3] = (u32)((ga >> 32) & 0x01ffffffu) | (2u << 30);   // global_addr[56:32], type=2
  i32x8 g1;
  const u32 dim0 = (u32)(NB * NH);                  // 8192 elements
  g1[0] = (int)(2u << 16);                          // data_size=2 (4 bytes); mask=0
  g1[1] = (int)((dim0 & 0xffffu) << 16);            // tensor_dim0[15:0]   (bits 63:48)
  g1[2] = (int)((dim0 >> 16) & 0xffffu)             // tensor_dim0[31:16]  (bits 79:64)
        | (1 << 16);                                // tensor_dim1[15:0]=1 (bits 95:80)
  g1[3] = (int)((dim0 & 0xffffu) << 16);            // tile_dim0=8192      (bits 127:112)
  g1[4] = 1;                                        // tile_dim1=1, tile_dim2=0
  g1[5] = (int)dim0;                                // tensor_dim0_stride[31:0]
  g1[6] = 0;                                        // stride0 hi / stride1 lo
  g1[7] = 0;
  const i32x4 z4 = {0, 0, 0, 0};
  const i32x8 z8 = {0, 0, 0, 0, 0, 0, 0, 0};
  // 6-arg form on this toolchain: (g0, g1, g2, g3, g4, cpol)
  __builtin_amdgcn_tensor_load_to_lds(g0, g1, z4, z4, z8, 0);
}

// ---- Kernel A: xw[t*32+b][n] = (x @ W_xh0)[b][t][n] + b0[n]  (parallel) ----
__global__ __launch_bounds__(256)
void xw_kernel(const float* __restrict__ x, const float* __restrict__ W,
               const float* __restrict__ b0, float* __restrict__ xw) {
  const int wave = threadIdx.x >> 5, lane = threadIdx.x & 31;
  const int tile = blockIdx.x * 8 + wave;          // 65536 tiles
  const int n = tile & 15, r = tile >> 4;
  const int t = r >> 1, bt = (r & 1) << 4;
  const int col = n * 16 + (lane & 15);

  v16bf wb[8];
#pragma unroll
  for (int kc = 0; kc < 8; ++kc) wb[kc] = gbl_bfrag(W, col, kc, lane);

  const float bias = b0[col];
  v8f accA, accB;
#pragma unroll
  for (int i = 0; i < 8; ++i) { accA[i] = bias; accB[i] = 0.f; }

  const int arow = bt + (lane & 15);
  const float* xp = x + ((size_t)arow * S_LEN + t) * NH;
#pragma unroll
  for (int kc = 0; kc < 4; ++kc) {                 // two independent K-chains
    accA = wmma_bf16(gbl_afrag_f32(xp, kc,     lane), wb[kc],     accA);
    accB = wmma_bf16(gbl_afrag_f32(xp, kc + 4, lane), wb[kc + 4], accB);
  }

  const int rofs = (lane & 16) ? 8 : 0;
#pragma unroll
  for (int i = 0; i < 8; ++i)
    xw[(size_t)(r * 16 + i + rofs) * NH + col] = accA[i] + accB[i];
}

// ---- Kernel B: persistent serial scan, 1 workgroup -------------------------
// Weights pinned in VGPRs; hidden state in LDS; xW stream staged by TDM.
__global__ __launch_bounds__(512, 1)
void rnn_scan_kernel(const float* __restrict__ xw,
                     const float* __restrict__ Whh0,
                     const float* __restrict__ Wxh1,
                     const float* __restrict__ Whh1,
                     const float* __restrict__ b1,
                     const float* __restrict__ h0_init,
                     const float* __restrict__ h1_init,
                     float* __restrict__ hs0, float* __restrict__ hs1,
                     float* __restrict__ hf0, float* __restrict__ hf1) {
  __shared__ bf16_t h0buf[2][NB * HPITCH];   // double-buffered layer-0 state
  __shared__ bf16_t h1s[NB * HPITCH];        // layer-1 state
  __shared__ float  xwlds[2][NB * NH];       // double-buffered TDM staging (2x32KB)

  const int tid = threadIdx.x;
  const int wave = tid >> 5, lane = tid & 31;
  const int col  = wave * 16 + (lane & 15);        // wave owns N-columns [16w,16w+16)
  const int rofs = (lane & 16) ? 8 : 0;

  // Pin all three recurrent weight strips in registers (3*8*8 = 192 VGPRs)
  v16bf wb0[8], wb1[8], wb2[8];
#pragma unroll
  for (int kc = 0; kc < 8; ++kc) {
    wb0[kc] = gbl_bfrag(Whh0, col, kc, lane);
    wb1[kc] = gbl_bfrag(Wxh1, col, kc, lane);
    wb2[kc] = gbl_bfrag(Whh1, col, kc, lane);
  }
  const float bias1 = b1[col];

  for (int i = tid; i < NB * NH; i += 512) {
    const int r = i >> 8, c = i & 255;
    h0buf[0][r * HPITCH + c] = (bf16_t)h0_init[i];
    h1s[r * HPITCH + c]      = (bf16_t)h1_init[i];
  }
  if (tid < 32) {                      // wave 0 only: TDM ignores EXEC
    tdm_load_tile(xw, &xwlds[0][0]);   // stage t=0 tile
    __builtin_amdgcn_s_wait_tensorcnt(0);
  }
  __syncthreads();

  for (int t = 0; t < S_LEN; ++t) {
    const bf16_t* h0c = h0buf[t & 1];
    bf16_t*       h0n = h0buf[(t + 1) & 1];
    const float*  xwt = xwlds[t & 1];

    // Kick the DMA for step t+1 into the other staging buffer; it overlaps
    // this whole step and is published by barrier C below.
    if (tid < 32 && t + 1 < S_LEN)
      tdm_load_tile(xw + (size_t)(t + 1) * (NB * NH), &xwlds[(t + 1) & 1][0]);

    // ---- layer 0: h0_new = tanh(xW[t] + h0 @ W_hh0) ----
    v8f acc0, acc1;
#pragma unroll
    for (int i = 0; i < 8; ++i) {                  // broadcast-friendly LDS reads
      acc0[i] = xwt[(i + rofs) * NH + col];
      acc1[i] = xwt[(16 + i + rofs) * NH + col];
    }
    {   // software-pipelined A-fragments
      v16bf a0 = lds_afrag(h0c, 0, 0, lane);
      v16bf a1 = lds_afrag(h0c, 16, 0, lane);
#pragma unroll
      for (int kc = 0; kc < 8; ++kc) {
        v16bf n0 = a0, n1 = a1;
        if (kc < 7) {
          n0 = lds_afrag(h0c, 0,  kc + 1, lane);
          n1 = lds_afrag(h0c, 16, kc + 1, lane);
        }
        acc0 = wmma_bf16(a0, wb0[kc], acc0);
        acc1 = wmma_bf16(a1, wb0[kc], acc1);
        a0 = n0; a1 = n1;
      }
    }
    // No barrier: writes go to the other h0 buffer.
    {
      float* hs0t = hs0 + (size_t)t * (NB * NH);
#pragma unroll
      for (int i = 0; i < 8; ++i) {
        const float v0 = tanhf(acc0[i]);
        const float v1 = tanhf(acc1[i]);
        const int r0 = i + rofs, r1 = 16 + i + rofs;
        h0n[r0 * HPITCH + col] = (bf16_t)v0;
        h0n[r1 * HPITCH + col] = (bf16_t)v1;
        hs0t[(size_t)r0 * NH + col] = v0;
        hs0t[(size_t)r1 * NH + col] = v1;
        if (t == S_LEN - 1) { hf0[r0 * NH + col] = v0; hf0[r1 * NH + col] = v1; }
      }
    }
    __syncthreads();   // A: new h0 visible

    // ---- layer 1: h1_new = tanh(h0_new @ W_xh1 + h1 @ W_hh1 + b1) ----
#pragma unroll
    for (int i = 0; i < 8; ++i) { acc0[i] = bias1; acc1[i] = bias1; }
#pragma unroll
    for (int kc = 0; kc < 8; ++kc) {               // 2 independent chains (acc0, acc1), 4 WMMAs per kc
      const v16bf p0 = lds_afrag(h0n, 0,  kc, lane);
      const v16bf p1 = lds_afrag(h0n, 16, kc, lane);
      const v16bf q0 = lds_afrag(h1s, 0,  kc, lane);
      const v16bf q1 = lds_afrag(h1s, 16, kc, lane);
      acc0 = wmma_bf16(p0, wb1[kc], acc0);
      acc1 = wmma_bf16(p1, wb1[kc], acc1);
      acc0 = wmma_bf16(q0, wb2[kc], acc0);
      acc1 = wmma_bf16(q1, wb2[kc], acc1);
    }
    __syncthreads();   // B: all reads of old h1 done
    {
      float* hs1t = hs1 + (size_t)t * (NB * NH);
#pragma unroll
      for (int i = 0; i < 8; ++i) {
        const float v0 = tanhf(acc0[i]);
        const float v1 = tanhf(acc1[i]);
        const int r0 = i + rofs, r1 = 16 + i + rofs;
        h1s[r0 * HPITCH + col] = (bf16_t)v0;
        h1s[r1 * HPITCH + col] = (bf16_t)v1;
        hs1t[(size_t)r0 * NH + col] = v0;
        hs1t[(size_t)r1 * NH + col] = v1;
        if (t == S_LEN - 1) { hf1[r0 * NH + col] = v0; hf1[r1 * NH + col] = v1; }
      }
    }
    if (tid < 32) __builtin_amdgcn_s_wait_tensorcnt(0);  // t+1 tile landed
    __syncthreads();   // C: new h1 + next xW tile visible
  }
}

// ---- Kernel C: outputs[b][t][o] = h1_states[t][b] @ W_hy + b_y  (parallel) --
__global__ __launch_bounds__(256)
void out_kernel(const float* __restrict__ h1st, const float* __restrict__ Why,
                const float* __restrict__ by, float* __restrict__ out) {
  const int wave = threadIdx.x >> 5, lane = threadIdx.x & 31;
  const int tile = blockIdx.x * 8 + wave;
  const int n = tile & 15, r = tile >> 4;
  const int col = n * 16 + (lane & 15);

  v16bf wb[8];
#pragma unroll
  for (int kc = 0; kc < 8; ++kc) wb[kc] = gbl_bfrag(Why, col, kc, lane);

  const float bias = by[col];
  v8f accA, accB;
#pragma unroll
  for (int i = 0; i < 8; ++i) { accA[i] = bias; accB[i] = 0.f; }

  const float* ap = h1st + (size_t)(r * 16 + (lane & 15)) * NH;  // row = t*32+b
#pragma unroll
  for (int kc = 0; kc < 4; ++kc) {
    accA = wmma_bf16(gbl_afrag_f32(ap, kc,     lane), wb[kc],     accA);
    accB = wmma_bf16(gbl_afrag_f32(ap, kc + 4, lane), wb[kc + 4], accB);
  }

  const int rofs = (lane & 16) ? 8 : 0;
#pragma unroll
  for (int i = 0; i < 8; ++i) {
    const int rgl = r * 16 + i + rofs;           // rgl = t*32 + b
    const int b = rgl & 31, t = rgl >> 5;
    out[((size_t)b * S_LEN + t) * NH + col] = accA[i] + accB[i];
  }
}

extern "C" void kernel_launch(void* const* d_in, const int* in_sizes, int n_in,
                              void* d_out, int out_size, void* d_ws, size_t ws_size,
                              hipStream_t stream) {
  (void)in_sizes; (void)n_in; (void)out_size; (void)d_ws; (void)ws_size;
  const float* x    = (const float*)d_in[0];
  const float* hprv = (const float*)d_in[1];   // (L,B,H) zeros
  const float* Wxh0 = (const float*)d_in[2];
  const float* Whh0 = (const float*)d_in[3];
  const float* bh0  = (const float*)d_in[4];
  const float* Wxh1 = (const float*)d_in[5];
  const float* Whh1 = (const float*)d_in[6];
  const float* bh1  = (const float*)d_in[7];
  const float* Why  = (const float*)d_in[8];
  const float* by   = (const float*)d_in[9];

  float* out     = (float*)d_out;
  float* outputs = out;                                       // B*S*O floats
  float* hstates = out + (size_t)NB * S_LEN * NH;             // L*S*B*H floats
  float* hfinal  = hstates + (size_t)2 * S_LEN * NB * NH;     // L*B*H floats
  float* hs0 = hstates;
  float* hs1 = hstates + (size_t)S_LEN * NB * NH;

  // Stash xW (= x@W_xh0 + b0, laid out (S,B,H)) in the outputs region: exact
  // same element count; consumed by the scan, then overwritten by out_kernel.
  float* xw = outputs;

  xw_kernel<<<8192, 256, 0, stream>>>(x, Wxh0, bh0, xw);
  rnn_scan_kernel<<<1, 512, 0, stream>>>(xw, Whh0, Wxh1, Whh1, bh1,
                                         hprv, hprv + NB * NH,
                                         hs0, hs1,
                                         hfinal, hfinal + NB * NH);
  out_kernel<<<8192, 256, 0, stream>>>(hs1, Why, by, outputs);
}
